// TensorAttention_56332791054854
// MI455X (gfx1250) — matrix operand order repaired and checked by reading, not compiled, not run on hardware
//
#include <hip/hip_runtime.h>
#include <hip/hip_bf16.h>

typedef __attribute__((ext_vector_type(16))) _Float16 v16h;
typedef __attribute__((ext_vector_type(4)))  _Float16 v4h;
typedef __attribute__((ext_vector_type(8)))  float    v8f;

#define B_    32
#define TD1   64
#define TD2   128
#define WD    256
#define PROJ  128
#define RH    64
#define ROWS_PER_WG 16
#define NTHREADS    128   // 4 wave32 waves

// ---- LDS layout (byte offsets; all 32B aligned) ----
#define OFF_WPROJ  0        // 64 B-frags * 1KB = 65536 B
#define OFF_WIN    65536    // 16 B-frags      -> 16384 B
#define OFF_WHID   81920    //  8 B-frags      ->  8192 B
#define OFF_XA0    90112    // x_t A-frags buf0+buf1 (2 x 8192 B, adjacent)
#define OFF_AA     106496   // a_t A-frags (4 frags)      -> 4096 B
#define OFF_HA     110592   // h   A-frags (2 frags)      -> 2048 B
#define OFF_SCORES 112640   // f32 16x128                 -> 8192 B
#define OFF_SCORER 120832   // f32 64                     ->  256 B
#define LDS_BYTES  121088
#define XA_HALFS   4096     // halfs per x A-frag buffer (8 frags * 512)

// hardware tanh if the toolchain exposes it for gfx1250
__device__ __forceinline__ float fast_tanh(float v) {
#if __has_builtin(__builtin_amdgcn_tanhf)
  return __builtin_amdgcn_tanhf(v);
#elif __has_builtin(__builtin_amdgcn_tanh_f32)
  return __builtin_amdgcn_tanh_f32(v);
#else
  return tanhf(v);
#endif
}

// A-matrix (16x32 f16) fragment offset for element (m, kk), kk = k % 32.
__device__ __forceinline__ int aoff(int m, int kk) {
  int L, e;
  if      (kk <  8) { L = m;      e = kk;      }
  else if (kk < 16) { L = m + 16; e = kk - 8;  }
  else if (kk < 24) { L = m;      e = kk - 8;  }
  else              { L = m + 16; e = kk - 16; }
  return L * 16 + e;
}

// coalesced registers<-global load of x(:, t, :) : 8 float4 per thread
__device__ __forceinline__ void load_xt(const float* __restrict__ x, int gr0,
                                        int t, int tid, float4* r) {
  #pragma unroll
  for (int q = 0; q < 8; ++q) {
    int qidx = q * NTHREADS + tid;
    int row  = qidx >> 6;            // 64 quads per 256-wide row
    int k    = (qidx & 63) * 4;
    r[q] = *(const float4*)(x + ((size_t)(gr0 + row) * TD2 + t) * WD + k);
  }
}

// drain registers -> LDS A-fragments (f16), 8-byte vector stores
__device__ __forceinline__ void store_xt(_Float16* s_xA, int tid, const float4* r) {
  #pragma unroll
  for (int q = 0; q < 8; ++q) {
    int qidx = q * NTHREADS + tid;
    int row  = qidx >> 6;
    int k    = (qidx & 63) * 4;
    v4h hv = { (_Float16)r[q].x, (_Float16)r[q].y, (_Float16)r[q].z, (_Float16)r[q].w };
    // k % 4 == 0 -> all 4 elements land consecutively inside one 8-half block
    *(v4h*)(s_xA + (k >> 5) * 512 + aoff(row, k & 31)) = hv;
  }
}

__global__ void __launch_bounds__(NTHREADS)
tensor_attention_fused(const float* __restrict__ x,
                       const float* __restrict__ att_proj,   // (WD, PROJ)
                       const float* __restrict__ w_in,       // (PROJ, RH)
                       const float* __restrict__ w_hid,      // (RH, RH)
                       const float* __restrict__ scorer,     // (RH,)
                       float* __restrict__ out)              // (B, TD1, WD)
{
  extern __shared__ __align__(64) char smem[];
  _Float16* s_wproj  = (_Float16*)(smem + OFF_WPROJ);
  _Float16* s_win    = (_Float16*)(smem + OFF_WIN);
  _Float16* s_whid   = (_Float16*)(smem + OFF_WHID);
  _Float16* s_xA0    = (_Float16*)(smem + OFF_XA0);   // buf = s_xA0 + b*XA_HALFS
  _Float16* s_aA     = (_Float16*)(smem + OFF_AA);
  _Float16* s_hA     = (_Float16*)(smem + OFF_HA);
  float*    s_scores = (float*)   (smem + OFF_SCORES);
  float*    s_scorer = (float*)   (smem + OFF_SCORER);

  const int tid  = threadIdx.x;
  const int lane = tid & 31;   // wave32
  const int wave = tid >> 5;   // 0..3
  const int gr0  = blockIdx.x * ROWS_PER_WG;

  // ---- preload weights as f16 in B-fragment swizzled order ----
  for (int idx = tid; idx < 64 * 512; idx += NTHREADS) {          // att_proj
    int f = idx >> 9, L = (idx >> 4) & 31, e = idx & 15;
    int nt = f >> 3, ki = f & 7;
    int n = nt * 16 + (L & 15);
    int k = ki * 32 + ((e >> 3) << 4) + ((L >> 4) << 3) + (e & 7);
    s_wproj[idx] = (_Float16)att_proj[k * PROJ + n];
  }
  for (int idx = tid; idx < 16 * 512; idx += NTHREADS) {          // w_in
    int f = idx >> 9, L = (idx >> 4) & 31, e = idx & 15;
    int nt = f >> 2, ki = f & 3;
    int n = nt * 16 + (L & 15);
    int k = ki * 32 + ((e >> 3) << 4) + ((L >> 4) << 3) + (e & 7);
    s_win[idx] = (_Float16)w_in[k * RH + n];
  }
  for (int idx = tid; idx < 8 * 512; idx += NTHREADS) {           // w_hid
    int f = idx >> 9, L = (idx >> 4) & 31, e = idx & 15;
    int nt = f >> 1, ki = f & 1;
    int n = nt * 16 + (L & 15);
    int k = ki * 32 + ((e >> 3) << 4) + ((L >> 4) << 3) + (e & 7);
    s_whid[idx] = (_Float16)w_hid[k * RH + n];
  }
  if (tid < RH) s_scorer[tid] = scorer[tid];
  for (int idx = tid; idx < 2 * 512; idx += NTHREADS) s_hA[idx] = (_Float16)0.f;
  for (int idx = tid; idx < ROWS_PER_WG * TD2; idx += NTHREADS) s_scores[idx] = 0.f;

  // prologue: stage x(:,0,:) into buffer 0
  {
    float4 r0[8];
    load_xt(x, gr0, 0, tid, r0);
    store_xt(s_xA0, tid, r0);
  }

  // ================= sequential scan over words (td2) =================
  for (int t = 0; t < TD2; ++t) {
    __syncthreads();                       // xA[cur] + weights visible
    _Float16* xcur = s_xA0 + (t & 1) * XA_HALFS;
    _Float16* xnxt = s_xA0 + ((t + 1) & 1) * XA_HALFS;

    // kick off next step's global loads early (hidden under WMMA work)
    float4 nx[8];
    if (t + 1 < TD2) load_xt(x, gr0, t + 1, tid, nx);

    // -- projection: a_t = tanh(x_t @ att_proj); 2 N-tiles per wave --
    for (int tt = 0; tt < 2; ++tt) {
      int nt = wave * 2 + tt;
      v8f acc = {};
      #pragma unroll
      for (int ki = 0; ki < 8; ++ki) {
        v16h av = *(const v16h*)(xcur + ki * 512 + lane * 16);
        v16h bv = *(const v16h*)(s_wproj + (nt * 8 + ki) * 512 + lane * 16);
        acc = __builtin_amdgcn_wmma_f32_16x16x32_f16(false, av, false, bv,
                                                     (short)0, acc, false, false);
      }
      int cm = (lane >> 4) * 8;            // C layout: M = r + 8*(lane/16)
      int cn = nt * 16 + (lane & 15);      //           N = lane % 16
      #pragma unroll
      for (int r = 0; r < 8; ++r) {
        float v = fast_tanh(acc[r]);
        s_aA[(cn >> 5) * 512 + aoff(cm + r, cn & 31)] = (_Float16)v;
      }
    }
    __syncthreads();

    // -- RNN step: h = tanh(a_t @ W_in + h_prev @ W_hid); 1 N-tile per wave --
    v16h hprev0 = *(const v16h*)(s_hA +       lane * 16);  // K 0..31
    v16h hprev1 = *(const v16h*)(s_hA + 512 + lane * 16);  // K 32..63
    __syncthreads();   // all reads of old h done before anyone writes new h
    {
      v8f acc = {};
      #pragma unroll
      for (int ki = 0; ki < 4; ++ki) {
        v16h av = *(const v16h*)(s_aA + ki * 512 + lane * 16);
        v16h bv = *(const v16h*)(s_win + (wave * 4 + ki) * 512 + lane * 16);
        acc = __builtin_amdgcn_wmma_f32_16x16x32_f16(false, av, false, bv,
                                                     (short)0, acc, false, false);
      }
      v16h bh0 = *(const v16h*)(s_whid + (wave * 2 + 0) * 512 + lane * 16);
      v16h bh1 = *(const v16h*)(s_whid + (wave * 2 + 1) * 512 + lane * 16);
      acc = __builtin_amdgcn_wmma_f32_16x16x32_f16(false, hprev0, false, bh0,
                                                   (short)0, acc, false, false);
      acc = __builtin_amdgcn_wmma_f32_16x16x32_f16(false, hprev1, false, bh1,
                                                   (short)0, acc, false, false);

      int cm = (lane >> 4) * 8;
      int cn = wave * 16 + (lane & 15);
      float scn = s_scorer[cn];
      #pragma unroll
      for (int r = 0; r < 8; ++r) {
        float v = fast_tanh(acc[r]);
        s_hA[(cn >> 5) * 512 + aoff(cm + r, cn & 31)] = (_Float16)v;
        // score: reduce h[m,n]*scorer[n] over the 16 lanes of this half-group
        float p = v * scn;
        #pragma unroll
        for (int off = 8; off >= 1; off >>= 1) p += __shfl_xor(p, off, 32);
        if ((lane & 15) == 0)
          atomicAdd(&s_scores[(cm + r) * TD2 + t], p);  // ds_add_f32, 4-way
      }
    }

    // drain prefetched x(:,t+1,:) into the other buffer
    if (t + 1 < TD2) store_xt(xnxt, tid, nx);
  }
  __syncthreads();

  // ---- softmax over words, in place ----
  if (tid < ROWS_PER_WG) {
    float mx = -1e30f;
    for (int t = 0; t < TD2; ++t) mx = fmaxf(mx, s_scores[tid * TD2 + t]);
    float sum = 0.f;
    for (int t = 0; t < TD2; ++t) {
      float e = __expf(s_scores[tid * TD2 + t] - mx);
      s_scores[tid * TD2 + t] = e;
      sum += e;
    }
    float inv = 1.f / sum;
    for (int t = 0; t < TD2; ++t) s_scores[tid * TD2 + t] *= inv;
  }
  __syncthreads();

  // ---- weighted sum of ORIGINAL x: out[row, w] = sum_t att * x (float4) ----
  for (int idx = tid; idx < ROWS_PER_WG * (WD / 4); idx += NTHREADS) {
    int row = idx >> 6;
    int w   = (idx & 63) * 4;
    const float* xr = x + (size_t)(gr0 + row) * TD2 * WD + w;
    float4 acc = make_float4(0.f, 0.f, 0.f, 0.f);
    for (int t = 0; t < TD2; ++t) {
      __builtin_prefetch(xr + (size_t)(t + 6) * WD, 0, 0);   // global_prefetch_b8
      float a = s_scores[row * TD2 + t];
      float4 xv = *(const float4*)(xr + (size_t)t * WD);
      acc.x += a * xv.x; acc.y += a * xv.y; acc.z += a * xv.z; acc.w += a * xv.w;
    }
    *(float4*)(out + (size_t)(gr0 + row) * WD + w) = acc;
  }
}

extern "C" void kernel_launch(void* const* d_in, const int* in_sizes, int n_in,
                              void* d_out, int out_size, void* d_ws, size_t ws_size,
                              hipStream_t stream) {
  (void)in_sizes; (void)n_in; (void)d_ws; (void)ws_size; (void)out_size;
  const float* x        = (const float*)d_in[0];
  const float* att_proj = (const float*)d_in[1];
  const float* w_in     = (const float*)d_in[2];
  const float* w_hid    = (const float*)d_in[3];
  const float* scorer   = (const float*)d_in[4];
  float* out = (float*)d_out;

  dim3 grid((B_ * TD1) / ROWS_PER_WG);   // 128 workgroups
  dim3 block(NTHREADS);                  // 4 wave32 waves
  hipLaunchKernelGGL(tensor_attention_fused, grid, block, LDS_BYTES, stream,
                     x, att_proj, w_in, w_hid, scorer, out);
}
